// EBSDSuper_51926154609193
// MI455X (gfx1250) — matrix operand order repaired and checked by reading, not compiled, not run on hardware
//
#include <hip/hip_runtime.h>

typedef __attribute__((ext_vector_type(16))) _Float16 v16h;
typedef __attribute__((ext_vector_type(8)))  float    v8f;

// ---- problem geometry (fixed by setup_inputs) ----
#define LR_W   128
#define N_LO   (128*128)       // 16384
#define HI_W   512
#define N_HI   (512*512)       // 262144
#define XPITCH 24              // padded f32 row pitch
#define KP     704             // 22 chunks * 32
#define OUT6_OFF (N_HI*9)

// ---- workspace byte offsets ----
#define OFF_C     0            // 13068 f32 Wigner tensors
#define OFF_WSP   52480        // 32*704 f16
#define OFF_WTP   97536        // 32*704 f16
#define OFF_MCTX  142592       // 32*128 f16
#define OFF_NEIGH 150784       // 16384*24 f32
#define OFF_X     1723648      // 16384*24 f32

// C tensor float offsets (layout [i][j][k], k fastest)
#define C444_O 0
#define C464_O 729
#define C644_O 1782
#define C664_O 2835
#define C446_O 4356
#define C466_O 5409
#define C646_O 6930
#define C666_O 8451
#define C424_O 10648
#define C624_O 11053
#define C426_O 11638
#define C626_O 12223

__device__ __forceinline__ int imin(int a, int b) { return a < b ? a : b; }

__device__ __forceinline__ unsigned packh2(float lo, float hi) {
  union { _Float16 h[2]; unsigned u; } z;
  z.h[0] = (_Float16)lo; z.h[1] = (_Float16)hi; return z.u;
}

struct U4x2 { uint4 a, b; };

__device__ __forceinline__ v16h frag2(const _Float16* p0, const _Float16* p1) {
  U4x2 u; u.a = *(const uint4*)p0; u.b = *(const uint4*)p1;
  return __builtin_bit_cast(v16h, u);
}

__device__ __forceinline__ v16h splat16(float v) {
  unsigned hv = packh2(v, v);
  uint4 q; q.x = hv; q.y = hv; q.z = hv; q.w = hv;
  U4x2 u; u.a = q; u.b = q;
  return __builtin_bit_cast(v16h, u);
}

__device__ __forceinline__ v8f wmma16(v16h a, v16h b, v8f c) {
  return __builtin_amdgcn_wmma_f32_16x16x32_f16(false, a, false, b, (short)0, c, false, false);
}

// same-wave LDS producer->consumer ordering (LDS ops are in-order per wave;
// the wait + memory clobber pins both HW completion and compiler ordering)
__device__ __forceinline__ void lds_fence() {
  asm volatile("s_wait_dscnt 0" ::: "memory");
}

__device__ __forceinline__ void load_row24(const float* base, float* a) {
  const float4* r = (const float4*)base;
#pragma unroll
  for (int q = 0; q < 6; q++) {
    float4 u = r[q];
    a[4*q] = u.x; a[4*q+1] = u.y; a[4*q+2] = u.z; a[4*q+3] = u.w;
  }
}

// pack 22 f32 -> 32 f16 (zero padded), 4x ds_store_b128
__device__ __forceinline__ void pack_row22(const float* v, _Float16* dst) {
  uint4* d = (uint4*)dst;
#pragma unroll
  for (int q = 0; q < 4; q++) {
    uint4 t;
    t.x = packh2((8*q+0 < 22) ? v[8*q+0] : 0.f, (8*q+1 < 22) ? v[8*q+1] : 0.f);
    t.y = packh2((8*q+2 < 22) ? v[8*q+2] : 0.f, (8*q+3 < 22) ? v[8*q+3] : 0.f);
    t.z = packh2((8*q+4 < 22) ? v[8*q+4] : 0.f, (8*q+5 < 22) ? v[8*q+5] : 0.f);
    t.w = packh2((8*q+6 < 22) ? v[8*q+6] : 0.f, (8*q+7 < 22) ? v[8*q+7] : 0.f);
    d[q] = t;
  }
}

// ================= setup: Wigner 3j tensors on device =================
__device__ const int W3J_TAB[12][4] = {
  {4,4,4,C444_O},{4,6,4,C464_O},{6,4,4,C644_O},{6,6,4,C664_O},
  {4,4,6,C446_O},{4,6,6,C466_O},{6,4,6,C646_O},{6,6,6,C666_O},
  {4,2,4,C424_O},{6,2,4,C624_O},{4,2,6,C426_O},{6,2,6,C626_O}};

__device__ double dfact(int n) { double r = 1.0; for (int i = 2; i <= n; i++) r *= (double)i; return r; }

__device__ double su2_cg(int j1,int m1,int j2,int m2,int j3,int m3) {
  if (m3 != m1 + m2) return 0.0;
  int vmin = -j1 + j2 + m3; if (-j1 + m1 > vmin) vmin = -j1 + m1; if (vmin < 0) vmin = 0;
  int vmax = j2 + j3 + m1; if (j3 - j1 + j2 < vmax) vmax = j3 - j1 + j2; if (j3 + m3 < vmax) vmax = j3 + m3;
  double c = sqrt((2.0*j3+1.0)*dfact(j3+j1-j2)*dfact(j3-j1+j2)*dfact(j1+j2-j3)/dfact(j1+j2+j3+1));
  c *= sqrt(dfact(j3+m3)*dfact(j3-m3)*dfact(j1-m1)*dfact(j1+m1)*dfact(j2-m2)*dfact(j2+m2));
  double s = 0.0;
  for (int v = vmin; v <= vmax; v++) {
    double sg = ((v + j2 + m2) & 1) ? -1.0 : 1.0;
    s += sg*dfact(j2+j3+m1-v)*dfact(j1-m1+v)/(dfact(v)*dfact(j3-j1+j2-v)*dfact(j3+m3-v)*dfact(v+j1-j2-m3));
  }
  return c * s;
}

// q(l)[a][b] including (-i)^l factor (l even -> real +-1)
__device__ void qent(int l, int a, int b, double& re, double& im) {
  const double s = 0.70710678118654752440;
  double r = 0.0, ii = 0.0;
  int mm = a - l;
  if (mm < 0)      { if (b == l - mm) r = s; else if (b == l + mm) ii = -s; }
  else if (mm == 0){ if (b == l) r = 1.0; }
  else             { double sg = (mm & 1) ? -1.0 : 1.0;
                     if (b == l + mm) r = sg*s; else if (b == l - mm) ii = sg*s; }
  double f = (l % 4 == 0) ? 1.0 : -1.0;
  re = r * f; im = ii * f;
}

__global__ void k_w3j(float* Cb) {
  __shared__ double cc[2197];
  __shared__ double red[256];
  const int b = blockIdx.x;
  const int l1 = W3J_TAB[b][0], l2 = W3J_TAB[b][1], l3 = W3J_TAB[b][2], off = W3J_TAB[b][3];
  const int n1 = 2*l1+1, n2 = 2*l2+1, n3 = 2*l3+1;
  const int tot = n1*n2*n3;
  for (int t = threadIdx.x; t < tot; t += 256) cc[t] = 0.0;
  __syncthreads();
  for (int t = threadIdx.x; t < n1*n2; t += 256) {
    int ai = t / n2, bi = t % n2;
    int m1 = ai - l1, m2 = bi - l2, m3 = m1 + m2;
    if (m3 >= -l3 && m3 <= l3) cc[(ai*n2+bi)*n3 + (l3+m3)] = su2_cg(l1,m1,l2,m2,l3,m3);
  }
  __syncthreads();
  double vals[9]; int cnt = 0; double ss = 0.0;
  for (int t = threadIdx.x; t < tot; t += 256) {
    int j = t/(n2*n3); int rem = t%(n2*n3); int lI = rem/n3; int mI = rem%n3;
    double are = 0.0;
    for (int i = 0; i < n1; i++) {
      double q1r,q1i; qent(l1,i,j,q1r,q1i);
      if (q1r == 0.0 && q1i == 0.0) continue;
      for (int k = 0; k < n2; k++) {
        double q2r,q2i; qent(l2,k,lI,q2r,q2i);
        if (q2r == 0.0 && q2i == 0.0) continue;
        double t12r = q1r*q2r - q1i*q2i, t12i = q1r*q2i + q1i*q2r;
        for (int n = 0; n < n3; n++) {
          double cv = cc[(i*n2+k)*n3+n];
          if (cv == 0.0) continue;
          double q3r,q3i; qent(l3,n,mI,q3r,q3i);  // conj(q3)
          are += cv*(t12r*q3r + t12i*q3i);
        }
      }
    }
    vals[cnt++] = are; ss += are*are;
  }
  red[threadIdx.x] = ss; __syncthreads();
  for (int s2 = 128; s2 > 0; s2 >>= 1) {
    if ((int)threadIdx.x < s2) red[threadIdx.x] += red[threadIdx.x + s2];
    __syncthreads();
  }
  double inv = 1.0 / sqrt(red[0]);
  cnt = 0;
  for (int t = threadIdx.x; t < tot; t += 256) Cb[off + t] = (float)(vals[cnt++] * inv);
}

// ============== setup: fold scalar weights into f16 B matrices ==============
__global__ void k_buildB(const float* __restrict__ Cb, const float* __restrict__ wsp,
                         const float* __restrict__ wtp, const float* __restrict__ wagg,
                         _Float16* __restrict__ Bsp, _Float16* __restrict__ Btp,
                         _Float16* __restrict__ Mctx) {
  int gid = blockIdx.x*256 + threadIdx.x;
  if (gid < 2*22528) {
    int which = gid / 22528; int e = gid % 22528;
    const float* w = which ? wtp : wsp;
    int n = e / KP; int k = e % KP; int i = k >> 5; int j = k & 31;
    float val = 0.f;
    if (n < 22 && i < 22 && j < 22) {
      if (n < 9) {
        if (i < 9) val = (j < 9) ? 1.5f*w[0]*Cb[C444_O+(i*9+j)*9+n]
                                 : 1.5f*w[1]*Cb[C464_O+(i*13+(j-9))*9+n];
        else       val = (j < 9) ? 1.5f*w[2]*Cb[C644_O+((i-9)*9+j)*9+n]
                                 : 1.5f*w[3]*Cb[C664_O+((i-9)*13+(j-9))*9+n];
      } else {
        int n6 = n - 9; const float s13 = 1.8027756377319946f;  // sqrt(13)/2
        if (i < 9) val = (j < 9) ? s13*w[4]*Cb[C446_O+(i*9+j)*13+n6]
                                 : s13*w[5]*Cb[C466_O+(i*13+(j-9))*13+n6];
        else       val = (j < 9) ? s13*w[6]*Cb[C646_O+((i-9)*9+j)*13+n6]
                                 : s13*w[7]*Cb[C666_O+((i-9)*13+(j-9))*13+n6];
      }
    }
    (which ? Btp : Bsp)[e] = (_Float16)val;
  } else if (gid < 2*22528 + 4096) {
    int e = gid - 2*22528;
    int n = e >> 7; int kk = e & 127; int p = kk >> 5; int i = kk & 31;
    float val = 0.f;
    if (n < 22 && i < 22) {
      const float s = 0.70710678118654752440f;
      float xx = (p & 1) ? s : -s;
      float yy = (p >> 1) ? s : -s;
      const float sh0 = 0.28209479177387814f;   // 0.5/sqrt(pi)
      const float kk2 = 1.0925484305920792f;    // 0.5*sqrt(15/pi)
      float y2[5];
      y2[0] = kk2*xx*yy; y2[1] = 0.f;
      y2[2] = -0.31539156525252005f;            // 0.25*sqrt(5/pi)*(3z^2-1), z=0
      y2[3] = 0.f; y2[4] = 0.5f*kk2*(xx*xx - yy*yy);
      float v = 0.f;
      if (n < 9) {
        if (i < 9) {
          if (i == n) v += wagg[0]*(1.f/3.f)*sh0;
          float s2 = 0.f;
#pragma unroll
          for (int j = 0; j < 5; j++) s2 += y2[j]*Cb[C424_O+(i*5+j)*9+n];
          v += wagg[1]*s2;
        } else {
          float s2 = 0.f;
#pragma unroll
          for (int j = 0; j < 5; j++) s2 += y2[j]*Cb[C624_O+((i-9)*5+j)*9+n];
          v += wagg[2]*s2;
        }
        v *= 1.7320508075688772f;               // sqrt(3)
      } else {
        int n6 = n - 9;
        if (i < 9) {
          float s2 = 0.f;
#pragma unroll
          for (int j = 0; j < 5; j++) s2 += y2[j]*Cb[C426_O+(i*5+j)*13+n6];
          v += wagg[3]*s2;
        } else {
          if (i - 9 == n6) v += wagg[4]*(1.f/3.6055512754639893f)*sh0;  // /sqrt(13)
          float s2 = 0.f;
#pragma unroll
          for (int j = 0; j < 5; j++) s2 += y2[j]*Cb[C626_O+((i-9)*5+j)*13+n6];
          v += wagg[5]*s2;
        }
        v *= 2.0816659994661326f;               // sqrt(13/3)
      }
      val = v;
    }
    Mctx[e] = (_Float16)val;
  }
}

// ================= pass 0: 3x3 edge-padded conv =================
__global__ void k_conv(const float* __restrict__ f4, const float* __restrict__ f6,
                       const float* __restrict__ sw, float* __restrict__ neigh) {
  int pix = blockIdx.x*256 + threadIdx.x;
  if (pix >= N_LO) return;
  int h = pix >> 7, w = pix & 127;
  float acc[22];
#pragma unroll
  for (int c = 0; c < 22; c++) acc[c] = 0.f;
  for (int di = 0; di < 3; di++) {
    int hh = h + di - 1; hh = hh < 0 ? 0 : (hh > 127 ? 127 : hh);
    for (int dj = 0; dj < 3; dj++) {
      int ww = w + dj - 1; ww = ww < 0 ? 0 : (ww > 127 ? 127 : ww);
      float wt = sw[di*3 + dj];
      int np = hh*LR_W + ww;
#pragma unroll
      for (int c = 0; c < 9; c++)  acc[c]   += wt*f4[np*9 + c];
#pragma unroll
      for (int c = 0; c < 13; c++) acc[9+c] += wt*f6[np*13 + c];
    }
  }
#pragma unroll
  for (int c = 0; c < 22; c++) neigh[pix*XPITCH + c] = acc[c];
  neigh[pix*XPITCH + 22] = 0.f; neigh[pix*XPITCH + 23] = 0.f;
}

// ================= pass 1: x = feats + tp_ff(feats, neigh, w_sp) =================
__global__ __launch_bounds__(256, 1)
void k_pass1(const float* __restrict__ f4, const float* __restrict__ f6,
             const float* __restrict__ neigh, const _Float16* __restrict__ Bg,
             float* __restrict__ x) {
  __shared__ __attribute__((aligned(16))) _Float16 sB[32*KP];       // 44 KB
  __shared__ __attribute__((aligned(16))) _Float16 sStage[8][16*32];// 8 KB
  {
    const uint4* src = (const uint4*)Bg; uint4* dst = (uint4*)sB;
    for (int t = threadIdx.x; t < 32*KP*2/16; t += 256) dst[t] = src[t];
  }
  __syncthreads();
  const int wave = threadIdx.x >> 5, lane = threadIdx.x & 31;
  const int m = lane & 15, half = lane >> 4;
  _Float16* stg = sStage[wave];
  const int waveId = blockIdx.x*8 + wave, stride = gridDim.x*8;
  for (int tile = waveId; tile < N_LO/16; tile += stride) {
    const int pix = tile*16 + m;
    float a[24];
#pragma unroll
    for (int c = 0; c < 9; c++)  a[c]   = f4[pix*9 + c];
#pragma unroll
    for (int c = 0; c < 13; c++) a[9+c] = f6[pix*13 + c];
    if (half == 0) {
      float nb[24]; load_row24(neigh + pix*XPITCH, nb);
      pack_row22(nb, stg + m*32);
    }
    lds_fence();
    const _Float16* bp = stg + m*32 + half*16;      // A-layout fragment of b-side
    v16h bfr = frag2(bp, bp + 8);
    v8f acc0 = {}, acc1 = {};
#pragma unroll
    for (int c = 0; c < 22; c++) {
      v16h af = bfr * splat16(a[c]);                // A[m][c*32+j] = a_c * b_j
      const _Float16* b0 = sB + m*KP + c*32 + half*16;
      acc0 = wmma16(af, frag2(b0, b0 + 8), acc0);
      const _Float16* b1 = sB + (16+m)*KP + c*32 + half*16;
      acc1 = wmma16(af, frag2(b1, b1 + 8), acc1);
    }
#pragma unroll
    for (int r = 0; r < 8; r++) {
      int mm = r + 8*half, p2 = tile*16 + mm;
      { int cn = m;
        float res = acc0[r] + ((cn < 9) ? f4[p2*9 + cn] : f6[p2*13 + (cn - 9)]);
        x[p2*XPITCH + cn] = res; }
      { int cn = 16 + m;
        if (cn < 22) { float res = acc1[r] + f6[p2*13 + (cn - 9)]; x[p2*XPITCH + cn] = res; } }
    }
    lds_fence();
  }
}

// ====== pass 2: context GEMM + out = ff + tp_ff(ff, context, w_tp) ======
__global__ __launch_bounds__(256, 1)
void k_pass2(const float* __restrict__ x, const _Float16* __restrict__ Bg,
             const _Float16* __restrict__ Mg, float* __restrict__ out) {
  __shared__ __attribute__((aligned(16))) _Float16 sB[32*KP];        // 44 KB
  __shared__ __attribute__((aligned(16))) _Float16 sM[32*128];       // 8 KB
  __shared__ __attribute__((aligned(16))) _Float16 sG[8][16*4*32];   // 32 KB
  __shared__ __attribute__((aligned(16))) _Float16 sC[8][16*32];     // 8 KB
  {
    const uint4* s1 = (const uint4*)Bg; uint4* d1 = (uint4*)sB;
    for (int t = threadIdx.x; t < 2816; t += 256) d1[t] = s1[t];
    const uint4* s2 = (const uint4*)Mg; uint4* d2 = (uint4*)sM;
    for (int t = threadIdx.x; t < 512; t += 256) d2[t] = s2[t];
  }
  __syncthreads();
  const int wave = threadIdx.x >> 5, lane = threadIdx.x & 31;
  const int m = lane & 15, half = lane >> 4;
  _Float16* gst = sG[wave]; _Float16* cst = sC[wave];
  const int waveId = blockIdx.x*8 + wave, stride = gridDim.x*8;
  for (int tile = waveId; tile < N_HI/16; tile += stride) {
    const int pixm = tile*16 + m;
    const int I = pixm >> 9, J = pixm & 511;
    // gather 2x2 neighborhood rows (edge clamp), stage as A-layout f16
#pragma unroll
    for (int e = 0; e < 2; e++) {
      int p = half*2 + e;
      int di = p >> 1, dj = p & 1;
      int Ic = imin(I + di, 511) >> 2, Jc = imin(J + dj, 511) >> 2;
      float nb[24]; load_row24(x + (Ic*LR_W + Jc)*XPITCH, nb);
      pack_row22(nb, gst + (m*4 + p)*32);
    }
    lds_fence();
    // context GEMM: ctx[m,k] = sum_{p,i} g[m][p*32+i] * Mctx[p][i][k]
    v8f c0 = {}, c1 = {};
#pragma unroll
    for (int p = 0; p < 4; p++) {
      const _Float16* ga = gst + (m*4 + p)*32 + half*16;
      v16h af = frag2(ga, ga + 8);
      const _Float16* b0 = sM + m*128 + p*32 + half*16;
      c0 = wmma16(af, frag2(b0, b0 + 8), c0);
      const _Float16* b1 = sM + (16+m)*128 + p*32 + half*16;
      c1 = wmma16(af, frag2(b1, b1 + 8), c1);
    }
    // D -> f16 ctx tile [16][32] (pads zero)
#pragma unroll
    for (int r = 0; r < 8; r++) {
      int mm = r + 8*half;
      cst[mm*32 + m] = (_Float16)c0[r];
      float v1 = (16 + m < 22) ? c1[r] : 0.f;
      cst[mm*32 + 16 + m] = (_Float16)v1;
    }
    lds_fence();
    // ff row (low-res parent of this pixel)
    const int low0 = (I >> 2)*LR_W + (J >> 2);
    float a[24]; load_row24(x + low0*XPITCH, a);
    const _Float16* cp = cst + m*32 + half*16;
    v16h bfr = frag2(cp, cp + 8);
    v8f acc0 = {}, acc1 = {};
#pragma unroll
    for (int c = 0; c < 22; c++) {
      v16h af = bfr * splat16(a[c]);
      const _Float16* b0 = sB + m*KP + c*32 + half*16;
      acc0 = wmma16(af, frag2(b0, b0 + 8), acc0);
      const _Float16* b1 = sB + (16+m)*KP + c*32 + half*16;
      acc1 = wmma16(af, frag2(b1, b1 + 8), acc1);
    }
    // residual + store (out4 | out6 concatenated)
#pragma unroll
    for (int r = 0; r < 8; r++) {
      int mm = r + 8*half, p2 = tile*16 + mm;
      int I2 = p2 >> 9, J2 = p2 & 511;
      int low2 = (I2 >> 2)*LR_W + (J2 >> 2);
      { int cn = m;
        float val = acc0[r] + x[low2*XPITCH + cn];
        if (cn < 9) out[p2*9 + cn] = val;
        else        out[OUT6_OFF + p2*13 + (cn - 9)] = val; }
      { int cn = 16 + m;
        if (cn < 22) {
          float val = acc1[r] + x[low2*XPITCH + cn];
          out[OUT6_OFF + p2*13 + (cn - 9)] = val; } }
    }
    lds_fence();
  }
}

extern "C" void kernel_launch(void* const* d_in, const int* in_sizes, int n_in,
                              void* d_out, int out_size, void* d_ws, size_t ws_size,
                              hipStream_t stream) {
  const float* f4   = (const float*)d_in[0];
  const float* f6   = (const float*)d_in[1];
  const float* sw   = (const float*)d_in[4];
  const float* wsp  = (const float*)d_in[5];
  const float* wagg = (const float*)d_in[6];
  const float* wtp  = (const float*)d_in[7];
  char* ws = (char*)d_ws;
  float*    Cb    = (float*)(ws + OFF_C);
  _Float16* Bsp   = (_Float16*)(ws + OFF_WSP);
  _Float16* Btp   = (_Float16*)(ws + OFF_WTP);
  _Float16* Mctx  = (_Float16*)(ws + OFF_MCTX);
  float*    neigh = (float*)(ws + OFF_NEIGH);
  float*    xbuf  = (float*)(ws + OFF_X);
  float*    out   = (float*)d_out;

  k_w3j   <<<dim3(12),  dim3(256), 0, stream>>>(Cb);
  k_buildB<<<dim3(192), dim3(256), 0, stream>>>(Cb, wsp, wtp, wagg, Bsp, Btp, Mctx);
  k_conv  <<<dim3(64),  dim3(256), 0, stream>>>(f4, f6, sw, neigh);
  k_pass1 <<<dim3(64),  dim3(256), 0, stream>>>(f4, f6, neigh, Bsp, xbuf);
  k_pass2 <<<dim3(256), dim3(256), 0, stream>>>(xbuf, Btp, Mctx, out);

  (void)in_sizes; (void)n_in; (void)out_size; (void)ws_size;
}
